// GraphClassifier_39840116638230
// MI455X (gfx1250) — matrix operand order repaired and checked by reading, not compiled, not run on hardware
//
#include <hip/hip_runtime.h>

// ---------------------------------------------------------------------------
// MI455X (gfx1250, wave32).
// fc1 dominates FLOPs (17.2 GFLOP) -> bf16 WMMA 16x16x32, f32 accumulate.
// Weights & inter-layer activations pre-packed in bf16 so the GEMM hot loop
// is pure load_b128 + v_wmma (no cvt): issue rate ~matches the XDL pipe.
// attention dominates bytes (268 MB adj/alpha) -> coalesced stream + ballot.
// ---------------------------------------------------------------------------

typedef __attribute__((ext_vector_type(16))) __bf16 v16bf;
typedef __attribute__((ext_vector_type(8)))  __bf16 v8bf;
typedef __attribute__((ext_vector_type(8)))  float  v8f;

#define N1 4096
#define N2 4096

// ---------------------------------------------------------------------------
// f32 -> packed bf16 (row-major layout preserved). Tiny one-shot kernels.
// ---------------------------------------------------------------------------
__global__ void cvt_f32_bf16(const float* __restrict__ in,
                             __bf16* __restrict__ out, int n) {
  int i = blockIdx.x * blockDim.x + threadIdx.x;
  if (i < n) out[i] = (__bf16)in[i];
}

// ---------------------------------------------------------------------------
// C[M,N] = act(A[M,K] @ W[N,K]^T + bias), N = NT*16.
// One wave per 16-row strip holds all NT accumulator tiles -> A streams from
// HBM exactly once; W (bf16, <=2MB) is L2-resident across row-blocks.
// gridDim.y = 2 selects encoder (doubles per-launch wave count; kernels on a
// stream serialize, so per-launch occupancy is what fills the WGPs).
// ---------------------------------------------------------------------------
struct GemmArgs {
  const void*   A;     // f32 (A_BF16=0) or bf16 (A_BF16=1), [M,K] row-major
  const __bf16* W;     // [N,K] row-major, packed bf16
  const float*  bias;  // [N]
  void*         C;     // f32 (OUT_BF16=0) or bf16 (OUT_BF16=1), [M,N]
};

template <int NT, bool A_BF16, bool RELU, bool OUT_BF16>
__global__ __launch_bounds__(32) void wmma_gemm(GemmArgs g0, GemmArgs g1, int K) {
  const GemmArgs g = (blockIdx.y == 0) ? g0 : g1;
  const int lane  = threadIdx.x;           // 0..31 (wave32)
  const int m0    = blockIdx.x * 16;
  const int rowA  = m0 + (lane & 15);
  const int kselA = (lane >> 4) * 16;      // 16-bit A frag: half-wave K split
  const int kselB = (lane >> 4) * 16;      // 16-bit B frag: half-wave K split
  const int Ncols = NT * 16;

  v8f acc[NT];
#pragma unroll
  for (int t = 0; t < NT; ++t) { v8f z = {}; acc[t] = z; }

  for (int kt = 0; kt < K; kt += 32) {
    // A fragment (16x32 bf16): lanes 0-15 hold K {0..15}, lanes 16-31
    // hold K {16..31}; vector elem e -> VGPR e/2, half e%2.
    v16bf a;
    if (A_BF16) {
      const __bf16* ap = (const __bf16*)g.A + (size_t)rowA * K + kt + kselA;
      v8bf lo = *(const v8bf*)(ap);
      v8bf hi = *(const v8bf*)(ap + 8);
      a = __builtin_shufflevector(lo, hi, 0, 1, 2, 3, 4, 5, 6, 7,
                                  8, 9, 10, 11, 12, 13, 14, 15);
    } else {
      const float* ap = (const float*)g.A + (size_t)rowA * K + kt + kselA;
#pragma unroll
      for (int e = 0; e < 8; ++e) {
        a[e]     = (__bf16)ap[e];
        a[8 + e] = (__bf16)ap[8 + e];
      }
    }
#pragma unroll
    for (int t = 0; t < NT; ++t) {
      // B fragment (32x16 bf16): lane -> column, VGPR v holds K = ksel+2v,+2v+1
      // 16 consecutive packed bf16 -> two clustered b128 loads, zero cvt.
      const int col = t * 16 + (lane & 15);
      v16bf b = *(const v16bf*)(g.W + (size_t)col * K + kt + kselB);
      acc[t] = __builtin_amdgcn_wmma_f32_16x16x32_bf16(
          false, a, false, b, (short)0, acc[t], false, false);
    }
  }

  // C/D layout: VGPR v -> row v (lanes 0-15) / row v+8 (lanes 16-31)
  const int cn   = lane & 15;
  const int mofs = (lane >> 4) * 8;
#pragma unroll
  for (int t = 0; t < NT; ++t) {
    const int col  = t * 16 + cn;
    const float bv = g.bias[col];
#pragma unroll
    for (int v = 0; v < 8; ++v) {
      float val = acc[t][v] + bv;
      if (RELU) val = fmaxf(val, 0.0f);
      const size_t idx = (size_t)(m0 + mofs + v) * Ncols + col;
      if (OUT_BF16)
        ((__bf16*)g.C)[idx] = (__bf16)val;
      else
        ((float*)g.C)[idx] = val;
    }
  }
}

// ---------------------------------------------------------------------------
// BatchNorm training-mode stats: per-column mean and 1/sqrt(var+eps).
// ---------------------------------------------------------------------------
__global__ __launch_bounds__(256) void bn_stats(const float* __restrict__ h,
                                                float* __restrict__ mean,
                                                float* __restrict__ inv,
                                                int rows) {
  __shared__ float ss[256], sq[256];
  const int c = blockIdx.x;  // 0..63
  float s = 0.f, q = 0.f;
  for (int r = threadIdx.x; r < rows; r += 256) {
    float v = h[(size_t)r * 64 + c];
    s += v;
    q += v * v;
  }
  ss[threadIdx.x] = s;
  sq[threadIdx.x] = q;
  __syncthreads();
  for (int off = 128; off > 0; off >>= 1) {
    if (threadIdx.x < off) {
      ss[threadIdx.x] += ss[threadIdx.x + off];
      sq[threadIdx.x] += sq[threadIdx.x + off];
    }
    __syncthreads();
  }
  if (threadIdx.x == 0) {
    float m = ss[0] / (float)rows;
    float v = sq[0] / (float)rows - m * m;
    mean[c] = m;
    inv[c]  = rsqrtf(fmaxf(v, 0.0f) + 1e-5f);
  }
}

__global__ void bn_apply_relu(const float* __restrict__ h,
                              const float* __restrict__ mean,
                              const float* __restrict__ inv,
                              const float* __restrict__ gamma,
                              const float* __restrict__ beta,
                              float* __restrict__ out, int n) {
  int i = blockIdx.x * blockDim.x + threadIdx.x;
  if (i < n) {
    int c = i & 63;
    float v = (h[i] - mean[c]) * inv[c] * gamma[c] + beta[c];
    out[i] = fmaxf(v, 0.0f);
  }
}

// ---------------------------------------------------------------------------
// Sparse attention row: out[i,:] = deg ? ((mask*alpha)[i,:] @ hd)/deg + hd[i,:] : 0
// One block (8 waves) per row; adj/alpha rows stream coalesced at full BW,
// nonzeros (~1%) broadcast via ballot/shfl; hd (1MB) is L2-resident.
// ---------------------------------------------------------------------------
__global__ __launch_bounds__(256) void attention_row(
    const float* __restrict__ adj, const float* __restrict__ alpha,
    const float* __restrict__ hd, float* __restrict__ out, int N) {
  const int i    = blockIdx.x;
  const int lane = threadIdx.x & 31;
  const int wave = threadIdx.x >> 5;  // 0..7
  const int span = N / 8;

  float2 acc = {0.f, 0.f};  // columns 2*lane, 2*lane+1
  float degp = 0.f;
  const float* arow = adj + (size_t)i * N;
  const float* lrow = alpha + (size_t)i * N;

  const int jend = (wave + 1) * span;
  for (int jb = wave * span; jb < jend; jb += 32) {
    float aj = arow[jb + lane];
    float al = lrow[jb + lane];
    degp += aj;
    unsigned m = (unsigned)__ballot(aj == 1.0f);
    while (m) {
      int b = __ffs(m) - 1;
      m &= m - 1;
      float w = __shfl(al, b, 32);
      float2 hv = ((const float2*)(hd + (size_t)(jb + b) * 64))[lane];
      acc.x += w * hv.x;
      acc.y += w * hv.y;
    }
  }
  for (int off = 16; off > 0; off >>= 1) degp += __shfl_down(degp, off, 32);

  __shared__ float sacc[8][64];
  __shared__ float sdeg[8];
  sacc[wave][2 * lane]     = acc.x;
  sacc[wave][2 * lane + 1] = acc.y;
  if (lane == 0) sdeg[wave] = degp;
  __syncthreads();

  if (threadIdx.x < 64) {
    const int c = threadIdx.x;
    float agg = 0.f, deg = 0.f;
#pragma unroll
    for (int w = 0; w < 8; ++w) { agg += sacc[w][c]; deg += sdeg[w]; }
    float v = (deg != 0.f) ? (agg / deg + hd[(size_t)i * 64 + c]) : 0.f;
    out[(size_t)i * 64 + c] = v;
  }
}

// ---------------------------------------------------------------------------
// Final classifier: logits = concat(att1,att2).flat @ Wc^T + bc (2 outputs).
// Two-stage deterministic reduction (no float atomics -> replay-stable).
// ---------------------------------------------------------------------------
__global__ __launch_bounds__(256) void cls_partial(
    const float* __restrict__ att1, const float* __restrict__ att2,
    const float* __restrict__ Wc, float* __restrict__ partial, int Khalf) {
  const int tid  = threadIdx.x;
  const int Ktot = 2 * Khalf;
  float p0 = 0.f, p1 = 0.f;
  for (int k = blockIdx.x * 256 + tid; k < Ktot; k += gridDim.x * 256) {
    float f = (k < Khalf) ? att1[k] : att2[k - Khalf];
    p0 += f * Wc[k];
    p1 += f * Wc[Ktot + k];
  }
  __shared__ float s0[256], s1[256];
  s0[tid] = p0;
  s1[tid] = p1;
  __syncthreads();
  for (int off = 128; off > 0; off >>= 1) {
    if (tid < off) { s0[tid] += s0[tid + off]; s1[tid] += s1[tid + off]; }
    __syncthreads();
  }
  if (tid == 0) {
    partial[blockIdx.x * 2]     = s0[0];
    partial[blockIdx.x * 2 + 1] = s1[0];
  }
}

__global__ __launch_bounds__(256) void cls_final(const float* __restrict__ partial,
                                                 const float* __restrict__ bc,
                                                 float* __restrict__ out, int nb) {
  __shared__ float s0[256], s1[256];
  float p0 = 0.f, p1 = 0.f;
  for (int b = threadIdx.x; b < nb; b += 256) {
    p0 += partial[2 * b];
    p1 += partial[2 * b + 1];
  }
  s0[threadIdx.x] = p0;
  s1[threadIdx.x] = p1;
  __syncthreads();
  for (int off = 128; off > 0; off >>= 1) {
    if (threadIdx.x < off) {
      s0[threadIdx.x] += s0[threadIdx.x + off];
      s1[threadIdx.x] += s1[threadIdx.x + off];
    }
    __syncthreads();
  }
  if (threadIdx.x == 0) {
    out[0] = s0[0] + bc[0];
    out[1] = s1[0] + bc[1];
  }
}

// ---------------------------------------------------------------------------
extern "C" void kernel_launch(void* const* d_in, const int* in_sizes, int n_in,
                              void* d_out, int out_size, void* d_ws, size_t ws_size,
                              hipStream_t stream) {
  (void)in_sizes; (void)n_in; (void)out_size; (void)ws_size;

  const float* x1    = (const float*)d_in[0];
  const float* x2    = (const float*)d_in[1];
  const float* adj1  = (const float*)d_in[2];
  const float* adj2  = (const float*)d_in[3];
  const float* e1W1  = (const float*)d_in[4];
  const float* e1b1  = (const float*)d_in[5];
  const float* e1W2  = (const float*)d_in[6];
  const float* e1b2  = (const float*)d_in[7];
  const float* e1W3  = (const float*)d_in[8];
  const float* e1b3  = (const float*)d_in[9];
  const float* e1g   = (const float*)d_in[10];
  const float* e1be  = (const float*)d_in[11];
  const float* e2W1  = (const float*)d_in[12];
  const float* e2b1  = (const float*)d_in[13];
  const float* e2W2  = (const float*)d_in[14];
  const float* e2b2  = (const float*)d_in[15];
  const float* e2W3  = (const float*)d_in[16];
  const float* e2b3  = (const float*)d_in[17];
  const float* e2g   = (const float*)d_in[18];
  const float* e2be  = (const float*)d_in[19];
  const float* alp1  = (const float*)d_in[20];
  const float* alp2  = (const float*)d_in[21];
  const float* Wc    = (const float*)d_in[22];
  const float* bc    = (const float*)d_in[23];
  float* out = (float*)d_out;

  // ---- workspace carve (f32 first, then bf16; ~17 MB total) ----
  float* pf = (float*)d_ws;
  float* h1c = pf; pf += (size_t)N1 * 64;   // fc3 out (f32, BN input)
  float* h2c = pf; pf += (size_t)N2 * 64;
  float* hd1 = pf; pf += (size_t)N1 * 64;   // BN+ReLU out
  float* hd2 = pf; pf += (size_t)N2 * 64;
  float* at1 = pf; pf += (size_t)N1 * 64;   // attention out
  float* at2 = pf; pf += (size_t)N2 * 64;
  float* mean1 = pf; pf += 64;
  float* inv1  = pf; pf += 64;
  float* mean2 = pf; pf += 64;
  float* inv2  = pf; pf += 64;
  float* partial = pf; pf += 512;

  __bf16* pb = (__bf16*)pf;
  __bf16* W1b1 = pb; pb += (size_t)256 * N1;   // weights, bf16-packed
  __bf16* W1b2 = pb; pb += (size_t)256 * N2;
  __bf16* W2b1 = pb; pb += (size_t)128 * 256;
  __bf16* W2b2 = pb; pb += (size_t)128 * 256;
  __bf16* W3b1 = pb; pb += (size_t)64 * 128;
  __bf16* W3b2 = pb; pb += (size_t)64 * 128;
  __bf16* h1a  = pb; pb += (size_t)N1 * 256;   // fc1 out (bf16 activations)
  __bf16* h2a  = pb; pb += (size_t)N2 * 256;
  __bf16* h1b  = pb; pb += (size_t)N1 * 128;   // fc2 out
  __bf16* h2b  = pb; pb += (size_t)N2 * 128;

  // ---- one-shot weight conversions to packed bf16 ----
  {
    int n;
    n = 256 * N1;  cvt_f32_bf16<<<(n + 255) / 256, 256, 0, stream>>>(e1W1, W1b1, n);
    n = 256 * N2;  cvt_f32_bf16<<<(n + 255) / 256, 256, 0, stream>>>(e2W1, W1b2, n);
    n = 128 * 256; cvt_f32_bf16<<<(n + 255) / 256, 256, 0, stream>>>(e1W2, W2b1, n);
    n = 128 * 256; cvt_f32_bf16<<<(n + 255) / 256, 256, 0, stream>>>(e2W2, W2b2, n);
    n = 64 * 128;  cvt_f32_bf16<<<(n + 255) / 256, 256, 0, stream>>>(e1W3, W3b1, n);
    n = 64 * 128;  cvt_f32_bf16<<<(n + 255) / 256, 256, 0, stream>>>(e2W3, W3b2, n);
  }

  const dim3 w32(32);
  const dim3 grid2(N1 / 16, 2);  // x = m-block, y = encoder select

  // fc1: [4096,4096] @ [256,4096]^T  (A f32 streamed once, W bf16, out bf16)
  {
    GemmArgs a0{x1, W1b1, e1b1, h1a};
    GemmArgs a1{x2, W1b2, e2b1, h2a};
    wmma_gemm<16, false, true, true><<<grid2, w32, 0, stream>>>(a0, a1, N1);
  }
  // fc2: K=256 -> N=128 (all-bf16 hot loop)
  {
    GemmArgs a0{h1a, W2b1, e1b2, h1b};
    GemmArgs a1{h2a, W2b2, e2b2, h2b};
    wmma_gemm<8, true, true, true><<<grid2, w32, 0, stream>>>(a0, a1, 256);
  }
  // fc3: K=128 -> N=64, no ReLU (BN follows), f32 out for BN stats
  {
    GemmArgs a0{h1b, W3b1, e1b3, h1c};
    GemmArgs a1{h2b, W3b2, e2b3, h2c};
    wmma_gemm<4, true, false, false><<<grid2, w32, 0, stream>>>(a0, a1, 128);
  }

  // BatchNorm (training stats) + ReLU
  bn_stats<<<64, 256, 0, stream>>>(h1c, mean1, inv1, N1);
  bn_stats<<<64, 256, 0, stream>>>(h2c, mean2, inv2, N2);
  bn_apply_relu<<<(N1 * 64 + 255) / 256, 256, 0, stream>>>(h1c, mean1, inv1, e1g, e1be, hd1, N1 * 64);
  bn_apply_relu<<<(N2 * 64 + 255) / 256, 256, 0, stream>>>(h2c, mean2, inv2, e2g, e2be, hd2, N2 * 64);

  // Sparse masked attention (memory-bound adj/alpha stream)
  attention_row<<<N1, 256, 0, stream>>>(adj1, alp1, hd1, at1, N1);
  attention_row<<<N2, 256, 0, stream>>>(adj2, alp2, hd2, at2, N2);

  // Final 2-logit classifier, deterministic two-stage reduction
  cls_partial<<<256, 256, 0, stream>>>(at1, at2, Wc, partial, N1 * 64);
  cls_final<<<1, 256, 0, stream>>>(partial, bc, out, 256);
}
